// PanODE_26371099197953
// MI455X (gfx1250) — matrix operand order repaired and checked by reading, not compiled, not run on hardware
//
#include <hip/hip_runtime.h>

typedef __attribute__((ext_vector_type(16))) _Float16 v16h;
typedef __attribute__((ext_vector_type(8)))  float    v8f;

#define BATCH 256
#define DIM   256
#define HID   256
#define NC    32      // num coeff
#define NB    30      // NC-2
#define NP    64      // collocation points
#define ITERS 20
#define PI_F  3.14159265358979323846f

// ---------------- workspace layout (bytes) ----------------
// consts (32KB):
//   f[0..63]          t_cheb
//   f[64..2111]       Phi_augT   [p=64][c=32]
//   f[2112..4031]     Phi_cT     [c=30][p=64]
//   f[4032..4091]     Phi_dT     [c=30][j=2]
//   f[4096..4415]     Tg_aug     [c=32][t=10]
// then: Aug f32 [256][32][256] (8MB), Yh f16 (8MB), Hh f16 (8MB),
//       Ft f32 (16MB), y16 f16 (128KB), W1t f16 (128KB), W2t f16 (128KB)

// Async 16B global -> LDS DMA (CDNA5, ASYNCcnt-tracked).
__device__ __forceinline__ void async_ld128(unsigned lds_addr, const void* gaddr) {
  asm volatile("global_load_async_to_lds_b128 %0, %1, off"
               :: "v"(lds_addr), "v"((unsigned long long)(uintptr_t)gaddr)
               : "memory");
}
__device__ __forceinline__ void wait_async0() {
  asm volatile("s_wait_asynccnt 0" ::: "memory");
}

// =====================================================================
// Small-matrix precompute (single block, 64 threads)
// =====================================================================
__global__ void pre_kernel(const float* __restrict__ tspan, int nt, float* __restrict__ C) {
  __shared__ float tch[NP];
  __shared__ float Phi[NC][NP], DPhi[NC][NP], Dt[NP];
  __shared__ float E[2][NP], G[2][NP];
  __shared__ float PhiB[NB][NP], PhiBT[NP][NB];
  __shared__ float S[NB][NB], Q[NB][NB];
  __shared__ float inv0[2][2], PhiTail[NB][2], R[2][NB];
  __shared__ float AW[NB][2 * NB];

  const int tid = threadIdx.x;
  const float t0 = tspan[0], t1 = tspan[nt - 1];
  const float sgn = (t1 > t0) ? 1.f : ((t1 < t0) ? -1.f : 0.f);

  if (tid < NP) {
    float tc = -sgn * cosf(PI_F * (float)tid / (float)NP);
    tch[tid] = tc;
    C[tid] = tc;
  }
  __syncthreads();

  if (tid < NP) {
    float tc = tch[tid];
    Dt[tid] = ((tid < NP - 1) ? tch[tid + 1] : 1.0f) - tc;
    Phi[0][tid] = 1.f; Phi[1][tid] = tc;
    for (int c = 2; c < NC; c++) Phi[c][tid] = 2.f * tc * Phi[c - 1][tid] - Phi[c - 2][tid];
    const float scale = 2.f / (t1 - t0);
    float up = 1.f, uc = 2.f * tc;           // U_0, U_1
    DPhi[0][tid] = 0.f;
    DPhi[1][tid] = scale * 1.f * up;         // 1 * U_0
    for (int c = 2; c < NC; c++) {
      DPhi[c][tid] = scale * (float)c * uc;  // c * U_{c-1}
      float un = 2.f * tc * uc - up; up = uc; uc = un;
    }
  }
  __syncthreads();

  if (tid == 0) {
    float a = Phi[0][0], b = DPhi[0][0], c2 = Phi[1][0], d = DPhi[1][0];
    float id = 1.f / (a * d - b * c2);
    inv0[0][0] =  d * id; inv0[0][1] = -b * id;
    inv0[1][0] = -c2 * id; inv0[1][1] =  a * id;
  }
  __syncthreads();

  if (tid < NB) { PhiTail[tid][0] = Phi[2 + tid][0]; PhiTail[tid][1] = DPhi[2 + tid][0]; }
  if (tid < NP) {
    for (int j = 0; j < 2; j++) {
      E[j][tid] = inv0[j][0] * DPhi[0][tid] + inv0[j][1] * DPhi[1][tid];
      G[j][tid] = inv0[j][0] * Phi[0][tid]  + inv0[j][1] * Phi[1][tid];
    }
  }
  __syncthreads();

  if (tid < NP) {
    float* PA = C + 64;
    PA[tid * NC + 0] = G[0][tid];
    PA[tid * NC + 1] = G[1][tid];
    for (int c = 0; c < NB; c++) {
      float pb = DPhi[2 + c][tid] - PhiTail[c][0] * E[0][tid] - PhiTail[c][1] * E[1][tid];
      PhiB[c][tid] = pb;
      PhiBT[tid][c] = Dt[tid] * pb;
      PA[tid * NC + 2 + c] = Phi[2 + c][tid] - PhiTail[c][0] * G[0][tid] - PhiTail[c][1] * G[1][tid];
    }
  }
  __syncthreads();

  if (tid < NB) {
    for (int c2i = 0; c2i < NB; c2i++) {
      float s = 0.f;
      for (int p = 0; p < NP; p++) s += PhiB[tid][p] * PhiBT[p][c2i];
      S[tid][c2i] = s;
    }
  }
  __syncthreads();

  if (tid == 0) {  // Gauss-Jordan inverse of S -> Q
    for (int i = 0; i < NB; i++)
      for (int j = 0; j < NB; j++) { AW[i][j] = S[i][j]; AW[i][NB + j] = (i == j) ? 1.f : 0.f; }
    for (int col = 0; col < NB; col++) {
      int piv = col; float mx = fabsf(AW[col][col]);
      for (int r2 = col + 1; r2 < NB; r2++) { float v = fabsf(AW[r2][col]); if (v > mx) { mx = v; piv = r2; } }
      if (piv != col) for (int j = 0; j < 2 * NB; j++) { float t = AW[col][j]; AW[col][j] = AW[piv][j]; AW[piv][j] = t; }
      float iv = 1.f / AW[col][col];
      for (int j = 0; j < 2 * NB; j++) AW[col][j] *= iv;
      for (int r2 = 0; r2 < NB; r2++) if (r2 != col) {
        float f = AW[r2][col];
        for (int j = 0; j < 2 * NB; j++) AW[r2][j] -= f * AW[col][j];
      }
    }
    for (int i = 0; i < NB; i++)
      for (int j = 0; j < NB; j++) Q[i][j] = AW[i][NB + j];
  }
  __syncthreads();

  if (tid < NP) {  // Phi_cT[c][p] = (Phi_bT @ Q)[p][c]
    for (int c = 0; c < NB; c++) {
      float s = 0.f;
      for (int c2i = 0; c2i < NB; c2i++) s += PhiBT[tid][c2i] * Q[c2i][c];
      C[2112 + c * NP + tid] = s;
    }
  }
  if (tid < NB) {  // R = E @ Phi_bT  (2 x 30)
    for (int j = 0; j < 2; j++) {
      float s = 0.f;
      for (int p = 0; p < NP; p++) s += E[j][p] * PhiBT[p][tid];
      R[j][tid] = s;
    }
  }
  __syncthreads();

  if (tid < NB) {  // Phi_dT[c][j] = (R @ Q)[j][c]
    for (int j = 0; j < 2; j++) {
      float s = 0.f;
      for (int c2i = 0; c2i < NB; c2i++) s += R[j][c2i] * Q[c2i][tid];
      C[4032 + tid * 2 + j] = s;
    }
  }
  if (tid == 0) {  // Tg_aug [32][nt]
    float* TA = C + 4096;
    for (int t = 0; t < nt; t++) {
      float to = -1.f + 2.f * (tspan[t] - t0) / (t1 - t0);
      float Tg[NC]; Tg[0] = 1.f; Tg[1] = to;
      for (int c = 2; c < NC; c++) Tg[c] = 2.f * to * Tg[c - 1] - Tg[c - 2];
      float g0 = inv0[0][0] * Tg[0] + inv0[0][1] * Tg[1];
      float g1 = inv0[1][0] * Tg[0] + inv0[1][1] * Tg[1];
      TA[0 * nt + t] = g0; TA[1 * nt + t] = g1;
      for (int c = 0; c < NB; c++)
        TA[(2 + c) * nt + t] = Tg[2 + c] - PhiTail[c][0] * g0 - PhiTail[c][1] * g1;
    }
  }
}

// =====================================================================
// Weight transpose + f16 convert:  Wt[n][k] = W[k][n]
// =====================================================================
__global__ void wconv_kernel(const float* __restrict__ W1, const float* __restrict__ W2,
                             _Float16* __restrict__ W1t, _Float16* __restrict__ W2t) {
  int r = blockIdx.x, c = threadIdx.x;
  W1t[r * 256 + c] = (_Float16)W1[c * 256 + r];
  W2t[r * 256 + c] = (_Float16)W2[c * 256 + r];
}

// =====================================================================
// State init: Aug[b][0][d]=y, Aug[b][2+c][d]=B_init[b][d][c], y16=f16(y)
// =====================================================================
__global__ void init_kernel(const float* __restrict__ y_init, const float* __restrict__ B_init,
                            float* __restrict__ Aug, _Float16* __restrict__ y16) {
  int b = blockIdx.x, d = threadIdx.x;
  float y = y_init[b * 256 + d];
  Aug[((size_t)b * NC + 0) * 256 + d] = y;
  y16[b * 256 + d] = (_Float16)y;
  const float* Bp = B_init + ((size_t)b * 256 + d) * NB;
  for (int c = 0; c < NB; c++) Aug[((size_t)b * NC + 2 + c) * 256 + d] = Bp[c];
}

__global__ void setf0_kernel(const float* __restrict__ F0, float* __restrict__ Aug) {
  int b = blockIdx.x, d = threadIdx.x;
  Aug[((size_t)b * NC + 1) * 256 + d] = F0[b * 256 + d];
}

// =====================================================================
// Y = Aug @ Phi_aug  ->  Yh[b][p][d] (f16)
// =====================================================================
__global__ __launch_bounds__(256) void y_kernel(const float* __restrict__ Aug,
                                                const float* __restrict__ PhiAugT,
                                                _Float16* __restrict__ Yh) {
  __shared__ float PA[NP * NC];
  int b = blockIdx.x, d = threadIdx.x;
  for (int i = d; i < NP * NC; i += 256) PA[i] = PhiAugT[i];
  float a[NC];
  const float* Ab = Aug + (size_t)b * NC * 256 + d;
#pragma unroll
  for (int c = 0; c < NC; c++) a[c] = Ab[c * 256];
  __syncthreads();
  for (int p = 0; p < NP; p++) {
    float s = 0.f;
#pragma unroll
    for (int c = 0; c < NC; c++) s += PA[p * NC + c] * a[c];
    Yh[((size_t)b * NP + p) * 256 + d] = (_Float16)s;
  }
}

// =====================================================================
// WMMA MLP GEMM: out = A[N,256] @ W (via Wt[n][k] f16), 64x256 tile/block
// Async-DMA staging, double-buffered W panel, 8 back-to-back WMMAs/wave.
// mode 1: tanh(acc + t*wt + b) -> f16   |  mode 2: acc + b -> f32
// =====================================================================
__global__ __launch_bounds__(256) void mlp_kernel(
    const _Float16* __restrict__ A, const _Float16* __restrict__ Wt,
    const float* __restrict__ bias, const float* __restrict__ twt,
    const float* __restrict__ tvals, int tmask, int mode,
    _Float16* __restrict__ outH, float* __restrict__ outF) {
  __shared__ __align__(16) _Float16 Alds[64 * 256];       // 32KB
  __shared__ __align__(16) _Float16 Wlds[2][256 * 32];    // 2 x 16KB panels [n][k]

  const int tid = threadIdx.x;
  const int rowbase = blockIdx.x * 64;
  const unsigned aBase = (unsigned)(uintptr_t)Alds;
  const unsigned wBase = (unsigned)(uintptr_t)&Wlds[0][0];

  // stage A tile (64 rows x 256 k): 8 x 16B DMA per thread
  {
    const char* gA = (const char*)(A + (size_t)rowbase * 256);
#pragma unroll
    for (int i = 0; i < 8; i++) {
      unsigned idx = (unsigned)tid + (unsigned)i * 256u;
      async_ld128(aBase + idx * 16u, gA + (size_t)idx * 16u);
    }
  }
  // stage W panel for k0=0 into buffer 0: 4 x 16B DMA per thread
#pragma unroll
  for (int i = 0; i < 4; i++) {
    int idx = tid + i * 256;
    int n = idx >> 2, q = idx & 3;
    async_ld128(wBase + (unsigned)idx * 16u, Wt + n * 256 + 0 + q * 8);
  }
  wait_async0();
  __syncthreads();

  const int lane = tid & 31, w = tid >> 5;
  const int rt = w & 3;              // row tile 0..3 (16 rows each)
  const int ctb = (w >> 2) * 8;      // 8 col tiles per wave
  v8f acc[8] = {};

  const int m_a  = lane & 15;
  const int kb_a = (lane >> 4) * 16;  // A frag K base (contiguous, matches B)
  const int n_b  = lane & 15;
  const int kb_b = (lane >> 4) * 16;  // B frag K base (contiguous, matches A)

  for (int k0 = 0; k0 < 256; k0 += 32) {
    const int buf = (k0 >> 5) & 1;

    // kick off DMA of next W panel into the other buffer (overlaps WMMAs)
    if (k0 + 32 < 256) {
      const int kn = k0 + 32;
#pragma unroll
      for (int i = 0; i < 4; i++) {
        int idx = tid + i * 256;
        int n = idx >> 2, q = idx & 3;
        async_ld128(wBase + 16384u * (unsigned)(buf ^ 1) + (unsigned)idx * 16u,
                    Wt + n * 256 + kn + q * 8);
      }
    }

    // A fragment (16x32 f16), contiguous k = kb_a + i per lane
    const _Float16* arow = Alds + (rt * 16 + m_a) * 256 + k0;
    v16h af;
#pragma unroll
    for (int i = 0; i < 16; i++) af[i] = arow[kb_a + i];

    // preload all 8 B fragments, then issue WMMAs back-to-back
    v16h bf[8];
#pragma unroll
    for (int ct = 0; ct < 8; ct++) {
      const _Float16* brow = &Wlds[buf][((ctb + ct) * 16 + n_b) * 32 + kb_b];
#pragma unroll
      for (int i = 0; i < 16; i++) bf[ct][i] = brow[i];
    }
#pragma unroll
    for (int ct = 0; ct < 8; ct++) {
      acc[ct] = __builtin_amdgcn_wmma_f32_16x16x32_f16(
          false, af, false, bf[ct], (short)0, acc[ct], false, false);
    }

    if (k0 + 32 < 256) {
      wait_async0();
      __syncthreads();
    }
  }

  // epilogue: C layout VGPR r: lanes0-15 M=r, lanes16-31 M=8+r; N=lane&15
#pragma unroll
  for (int ct = 0; ct < 8; ct++) {
    int col = (ctb + ct) * 16 + (lane & 15);
    float bcol = bias[col];
    float wcol = (mode == 1) ? twt[col] : 0.f;
#pragma unroll
    for (int r = 0; r < 8; r++) {
      int m = r + ((lane >= 16) ? 8 : 0);
      int row = rowbase + rt * 16 + m;
      float v = acc[ct][r];
      if (mode == 1) {
        float t = tvals[row & tmask];
        v = tanhf(v + t * wcol + bcol);
        outH[(size_t)row * 256 + col] = (_Float16)v;
      } else {
        outF[(size_t)row * 256 + col] = v + bcol;
      }
    }
  }
}

// =====================================================================
// Update: Aug[b][2+c][d] = sum_p PhiCT[c][p]*Ft[b][p][d] - PhiDT[c][:]·yf
// =====================================================================
__global__ __launch_bounds__(256) void upd_kernel(const float* __restrict__ Ft,
                                                  const float* __restrict__ PhiCT,
                                                  const float* __restrict__ PhiDT,
                                                  float* __restrict__ Aug) {
  __shared__ float PC[NB * NP];
  __shared__ float PD[NB * 2];
  int b = blockIdx.x, d = threadIdx.x;
  for (int i = d; i < NB * NP; i += 256) PC[i] = PhiCT[i];
  if (d < NB * 2) PD[d] = PhiDT[d];
  __syncthreads();
  float acc[NB];
#pragma unroll
  for (int c = 0; c < NB; c++) acc[c] = 0.f;
  const float* Fb = Ft + (size_t)b * NP * 256 + d;
  for (int p = 0; p < NP; p++) {
    float f = Fb[p * 256];
#pragma unroll
    for (int c = 0; c < NB; c++) acc[c] += PC[c * NP + p] * f;
  }
  float* Ab = Aug + (size_t)b * NC * 256 + d;
  float y0 = Ab[0], y1 = Ab[256];
#pragma unroll
  for (int c = 0; c < NB; c++) Ab[(2 + c) * 256] = acc[c] - PD[c * 2] * y0 - PD[c * 2 + 1] * y1;
}

// =====================================================================
// traj[t,b,d] = sum_c Aug[b][c][d] * TgAug[c][t]; out[0..nt)=t_span
// =====================================================================
__global__ __launch_bounds__(256) void traj_kernel(const float* __restrict__ Aug,
                                                   const float* __restrict__ TgAug,
                                                   const float* __restrict__ tspan, int nt,
                                                   float* __restrict__ out) {
  __shared__ float TA[NC * 16];
  int b = blockIdx.x, d = threadIdx.x;
  for (int i = d; i < NC * nt; i += 256) TA[i] = TgAug[i];
  float a[NC];
  const float* Ab = Aug + (size_t)b * NC * 256 + d;
#pragma unroll
  for (int c = 0; c < NC; c++) a[c] = Ab[c * 256];
  __syncthreads();
  for (int t = 0; t < nt; t++) {
    float s = 0.f;
#pragma unroll
    for (int c = 0; c < NC; c++) s += a[c] * TA[c * nt + t];
    out[nt + ((size_t)t * BATCH + b) * DIM + d] = s;
  }
  if (b == 0 && d < nt) out[d] = tspan[d];
}

// =====================================================================
extern "C" void kernel_launch(void* const* d_in, const int* in_sizes, int n_in,
                              void* d_out, int out_size, void* d_ws, size_t ws_size,
                              hipStream_t stream) {
  const float* y_init = (const float*)d_in[0];
  const float* t_span = (const float*)d_in[1];
  const float* B_init = (const float*)d_in[2];
  const float* W1     = (const float*)d_in[3];
  const float* b1     = (const float*)d_in[4];
  const float* wt     = (const float*)d_in[5];
  const float* W2     = (const float*)d_in[6];
  const float* b2     = (const float*)d_in[7];
  float* out = (float*)d_out;
  const int nt = in_sizes[1];

  char* ws = (char*)d_ws;
  float*    C   = (float*)ws;                                       // 32KB consts
  float*    Aug = (float*)(ws + (32u << 10));                       // 8MB
  _Float16* Yh  = (_Float16*)(ws + (32u << 10) + 1u * 8388608u);    // 8MB
  _Float16* Hh  = (_Float16*)(ws + (32u << 10) + 2u * 8388608u);    // 8MB
  float*    Ft  = (float*)(ws + (32u << 10) + 3u * 8388608u);       // 16MB
  char*     tail = ws + (32u << 10) + 3u * 8388608u + 16777216u;
  _Float16* y16 = (_Float16*)(tail);                                // 128KB
  _Float16* W1t = (_Float16*)(tail + 131072u);                      // 128KB
  _Float16* W2t = (_Float16*)(tail + 262144u);                      // 128KB

  pre_kernel<<<1, 64, 0, stream>>>(t_span, nt, C);
  wconv_kernel<<<256, 256, 0, stream>>>(W1, W2, W1t, W2t);
  init_kernel<<<256, 256, 0, stream>>>(y_init, B_init, Aug, y16);

  // f0 = f(t0, y_init) via the same WMMA MLP (N = 256 rows)
  mlp_kernel<<<BATCH / 64, 256, 0, stream>>>(y16, W1t, b1, wt, t_span, 0, 1, Hh, nullptr);
  mlp_kernel<<<BATCH / 64, 256, 0, stream>>>(Hh, W2t, b2, nullptr, nullptr, 0, 2, nullptr, Ft);
  setf0_kernel<<<256, 256, 0, stream>>>(Ft, Aug);

  const int NROWS = BATCH * NP;   // 16384
  for (int it = 0; it < ITERS; it++) {
    y_kernel<<<256, 256, 0, stream>>>(Aug, C + 64, Yh);
    mlp_kernel<<<NROWS / 64, 256, 0, stream>>>(Yh, W1t, b1, wt, C /*t_cheb*/, 63, 1, Hh, nullptr);
    mlp_kernel<<<NROWS / 64, 256, 0, stream>>>(Hh, W2t, b2, nullptr, nullptr, 0, 2, nullptr, Ft);
    upd_kernel<<<256, 256, 0, stream>>>(Ft, C + 2112, C + 4032, Aug);
  }

  traj_kernel<<<256, 256, 0, stream>>>(Aug, C + 4096, t_span, nt, out);
}
